// DiscoveryNet_53498112639027
// MI455X (gfx1250) — matrix operand order repaired and checked by reading, not compiled, not run on hardware
//
#include <hip/hip_runtime.h>

typedef __attribute__((ext_vector_type(16))) _Float16 v16h;
typedef __attribute__((ext_vector_type(8)))  float    v8f;
typedef __attribute__((ext_vector_type(2)))  float    v2f;

#define N_PTS  512
#define HID    128
#define DIMS   3
#define WAVES_PER_BLOCK 4
#define BLOCK  (WAVES_PER_BLOCK * 32)
#define ROWP   136          // padded LDS row stride in halfs (272B, 16B-aligned, bank-skewed)
#define TILE_J 16

// Single tanh (used in scalar spots). v_nop covers the TRANS-op result hazard.
__device__ __forceinline__ float fast_tanh(float x) {
    float r;
    asm("v_tanh_f32 %0, %1\n\tv_nop" : "=v"(r) : "v"(x));
    return r;
}

// Batched tanh: 8 independent back-to-back v_tanh_f32; each trans op is the
// "1 independent instruction" hazard cover for the previous one, so only a
// single trailing v_nop is needed for the whole group. Early-clobber outputs
// keep the inputs live for the silu reconstruction fma.
__device__ __forceinline__ void tanh8(float t[8], const float h[8]) {
    asm("v_tanh_f32 %0, %8\n\t"
        "v_tanh_f32 %1, %9\n\t"
        "v_tanh_f32 %2, %10\n\t"
        "v_tanh_f32 %3, %11\n\t"
        "v_tanh_f32 %4, %12\n\t"
        "v_tanh_f32 %5, %13\n\t"
        "v_tanh_f32 %6, %14\n\t"
        "v_tanh_f32 %7, %15\n\t"
        "v_nop"
        : "=&v"(t[0]), "=&v"(t[1]), "=&v"(t[2]), "=&v"(t[3]),
          "=&v"(t[4]), "=&v"(t[5]), "=&v"(t[6]), "=&v"(t[7])
        : "v"(h[0]), "v"(h[1]), "v"(h[2]), "v"(h[3]),
          "v"(h[4]), "v"(h[5]), "v"(h[6]), "v"(h[7]));
}

__global__ void __launch_bounds__(BLOCK) pairforce_zero_kernel(float* out, int n) {
    int idx = blockIdx.x * blockDim.x + threadIdx.x;
    if (idx < n) out[idx] = 0.0f;
}

__global__ void __launch_bounds__(BLOCK) pairforce_wmma_kernel(
    const float* __restrict__ pos, const float* __restrict__ W1,
    const float* __restrict__ b1,  const float* __restrict__ W2,
    const float* __restrict__ b2,  const float* __restrict__ W3,
    const float* __restrict__ b3,  float* __restrict__ out, int Bn)
{
    // W2 stored transposed AND pre-scaled by 0.5 (folds the silu half-argument):
    // sW2T[n][k] = 0.5*W2[k][n]  -> B-fragment loads contiguous in K
    __shared__ _Float16 sW2T[HID][ROWP];
    __shared__ _Float16 sH1[WAVES_PER_BLOCK][16][ROWP];
    __shared__ float sMag[WAVES_PER_BLOCK][16];

    const int tid = threadIdx.x;
    for (int idx = tid; idx < HID * HID; idx += BLOCK) {
        int k = idx >> 7, n = idx & 127;
        sW2T[n][k] = (_Float16)(0.5f * W2[idx]);
    }
    __syncthreads();

    const int wv   = tid >> 5;
    const int lane = tid & 31;
    const int m16  = lane & 15;   // C/D column (N); this lane's pair row in A
    const int half = lane >> 4;

    // ---- per-lane, tile-invariant constants hoisted out of the tile loop ----
    // Layer-1 B fragment carries the bias in the K=2 slot (lanes 16-31):
    //   B rows = [W1[0][n], W1[1][n], b1[n], 0]
    float w1r0[8], w1r1[8], hb2n[8], w3n[8];
    #pragma unroll
    for (int nt = 0; nt < 8; ++nt) {
        const int n = nt * 16 + m16;
        w1r0[nt] = (half == 0) ? W1[n]       : b1[n];   // K=0 row | K=2 row (bias)
        w1r1[nt] = (half == 0) ? W1[HID + n] : 0.0f;    // K=1 row | K=3 row (pad)
        hb2n[nt] = 0.5f * b2[n];                        // pre-halved b2
        w3n[nt]  = W3[n];
    }
    const float bias3 = b3[0];

    const int tilesPerBatch = N_PTS * (N_PTS / TILE_J);
    const int totalTiles    = Bn * tilesPerBatch;
    const int gwave  = blockIdx.x * WAVES_PER_BLOCK + wv;
    const int nwaves = gridDim.x * WAVES_PER_BLOCK;

    for (int t = gwave; t < totalTiles; t += nwaves) {
        const int b   = t / tilesPerBatch;
        const int rem = t - b * tilesPerBatch;
        const int i   = rem >> 5;            // N_PTS/TILE_J = 32 tiles per row
        const int j0  = (rem & 31) << 4;

        const float* pb = pos + (size_t)b * N_PTS * DIMS;
        const float xi = pb[i * 3 + 0], yi = pb[i * 3 + 1], zi = pb[i * 3 + 2];
        const int   j  = j0 + m16;
        const float dx = xi - pb[j * 3 + 0];
        const float dy = yi - pb[j * 3 + 1];
        const float dz = zi - pb[j * 3 + 2];
        const float s2   = dx * dx + dy * dy + dz * dz;
        const float dist = __builtin_amdgcn_sqrtf(s2);          // sqrt(0)=0 on diagonal
        const float dsafe = fminf(fmaxf(dist, 1e-4f), 50.0f);
        const float f0 = dsafe;
        const float f1 = __builtin_amdgcn_rcpf(dsafe);

#if __has_builtin(__builtin_amdgcn_wmma_f32_16x16x4_f32)
        // ---- layer 1 as fp32 WMMA: [0.5f0, 0.5f1, 0.5, 0] @ [W1row0; W1row1; b1; 0]
        //      => d1 = 0.5*(feat@W1 + b1) = silu half-argument h, no epilogue add/mul
        v2f afeat;
        afeat[0] = (half == 0) ? 0.5f * f0 : 0.5f;   // K=0 | K=2 (bias multiplier)
        afeat[1] = (half == 0) ? 0.5f * f1 : 0.0f;   // K=1 | K=3 (pad)
        v8f d1[8];
        #pragma unroll
        for (int nt = 0; nt < 8; ++nt) {
            v2f bf;
            bf[0] = w1r0[nt];
            bf[1] = w1r1[nt];
            v8f z = {};
            d1[nt] = __builtin_amdgcn_wmma_f32_16x16x4_f32(
                false, afeat, false, bf, (short)0, z, false, false);
        }
        // epilogue: silu(2h) = h*tanh(h)+h, f16, scatter into A-layout LDS tile
        #pragma unroll
        for (int nt = 0; nt < 8; ++nt) {
            float h[8], tt[8];
            #pragma unroll
            for (int r = 0; r < 8; ++r) h[r] = d1[nt][r];
            tanh8(tt, h);
            #pragma unroll
            for (int r = 0; r < 8; ++r)
                sH1[wv][r + 8 * half][nt * 16 + m16] = (_Float16)fmaf(h[r], tt[r], h[r]);
        }
#else
        // ---- fallback: VALU layer 1, k-contiguous writes ----
        {
            _Float16* h1row = &sH1[wv][m16][0];
            const int k0 = half * 64;
            #pragma unroll
            for (int kk = 0; kk < 64; kk += 2) {
                const int k = k0 + kk;
                float h0 = 0.5f * fmaf(f0, W1[k],     fmaf(f1, W1[HID + k],     b1[k]));
                float h1 = 0.5f * fmaf(f0, W1[k + 1], fmaf(f1, W1[HID + k + 1], b1[k + 1]));
                union { _Float16 h[2]; unsigned u; } pk;
                pk.h[0] = (_Float16)fmaf(h0, fast_tanh(h0), h0);
                pk.h[1] = (_Float16)fmaf(h1, fast_tanh(h1), h1);
                *(unsigned*)(h1row + k) = pk.u;
            }
        }
#endif
        asm volatile("s_wait_dscnt 0x0" ::: "memory");

        // ---- layer 2: 16x128 @ 128x128 via 32x v_wmma_f32_16x16x32_f16 ----
        const _Float16* ap = &sH1[wv][m16][0];
        v8f acc[8];
        {   // kc = 0 peeled: C = inline 0 (no accumulator init movs)
            v16h afrag;
            ((float4*)&afrag)[0] = *(const float4*)(ap + half * 8);
            ((float4*)&afrag)[1] = *(const float4*)(ap + half * 8 + 16);
            #pragma unroll
            for (int nt = 0; nt < 8; ++nt) {
                v16h bfrag;
                const _Float16* bp = &sW2T[nt * 16 + m16][half * 8];
                ((float4*)&bfrag)[0] = *(const float4*)(bp);
                ((float4*)&bfrag)[1] = *(const float4*)(bp + 16);
                v8f z = {};
                acc[nt] = __builtin_amdgcn_wmma_f32_16x16x32_f16(
                    false, afrag, false, bfrag, (short)0, z, false, false);
            }
        }
        #pragma unroll
        for (int kc = 1; kc < 4; ++kc) {
            v16h afrag;
            const int kA0 = kc * 32 + half * 8;
            ((float4*)&afrag)[0] = *(const float4*)(ap + kA0);
            ((float4*)&afrag)[1] = *(const float4*)(ap + kA0 + 16);
            #pragma unroll
            for (int nt = 0; nt < 8; ++nt) {
                v16h bfrag;
                const _Float16* bp = &sW2T[nt * 16 + m16][kc * 32 + half * 8];
                ((float4*)&bfrag)[0] = *(const float4*)(bp);
                ((float4*)&bfrag)[1] = *(const float4*)(bp + 16);
                acc[nt] = __builtin_amdgcn_wmma_f32_16x16x32_f16(
                    false, afrag, false, bfrag, (short)0, acc[nt], false, false);
            }
        }

        // ---- layer 3: acc is already 0.5*(h1@W2); h = acc + 0.5*b2, silu = h*tanh(h)+h
        float magp[8];
        #pragma unroll
        for (int r = 0; r < 8; ++r) magp[r] = 0.0f;
        #pragma unroll
        for (int nt = 0; nt < 8; ++nt) {
            float h[8], tt[8];
            #pragma unroll
            for (int r = 0; r < 8; ++r) h[r] = acc[nt][r] + hb2n[nt];
            tanh8(tt, h);
            #pragma unroll
            for (int r = 0; r < 8; ++r)
                magp[r] = fmaf(fmaf(h[r], tt[r], h[r]), w3n[nt], magp[r]);
        }
        // reduce over the 16 columns held by the lanes of each half
        #pragma unroll
        for (int r = 0; r < 8; ++r) {
            float v = magp[r];
            v += __shfl_xor(v, 1, 32);
            v += __shfl_xor(v, 2, 32);
            v += __shfl_xor(v, 4, 32);
            v += __shfl_xor(v, 8, 32);
            magp[r] = v;   // half 0: rows 0..7 ; half 1: rows 8..15
        }
        if (lane == 0) {
            #pragma unroll
            for (int r = 0; r < 8; ++r) sMag[wv][r] = magp[r];
        }
        if (lane == 16) {
            #pragma unroll
            for (int r = 0; r < 8; ++r) sMag[wv][8 + r] = magp[r];
        }
        asm volatile("s_wait_dscnt 0x0" ::: "memory");

        // ---- pair forces + reduction over the 16 j's (lanes 0..15 active) ----
        float fx = 0.0f, fy = 0.0f, fz = 0.0f;
        if (half == 0) {
            const float mag = sMag[wv][m16] + bias3;
            const float g   = mag * __builtin_amdgcn_rcpf(fmaxf(dist, 1e-6f)); // diagonal: diff==0 -> 0
            fx = g * dx; fy = g * dy; fz = g * dz;
        }
        fx += __shfl_xor(fx, 1, 32); fy += __shfl_xor(fy, 1, 32); fz += __shfl_xor(fz, 1, 32);
        fx += __shfl_xor(fx, 2, 32); fy += __shfl_xor(fy, 2, 32); fz += __shfl_xor(fz, 2, 32);
        fx += __shfl_xor(fx, 4, 32); fy += __shfl_xor(fy, 4, 32); fz += __shfl_xor(fz, 4, 32);
        fx += __shfl_xor(fx, 8, 32); fy += __shfl_xor(fy, 8, 32); fz += __shfl_xor(fz, 8, 32);
        if (lane == 0) {
            float* op = out + ((size_t)b * N_PTS + i) * 3;
            atomicAdd(op + 0, fx);
            atomicAdd(op + 1, fy);
            atomicAdd(op + 2, fz);
        }
    }
}

extern "C" void kernel_launch(void* const* d_in, const int* in_sizes, int n_in,
                              void* d_out, int out_size, void* d_ws, size_t ws_size,
                              hipStream_t stream) {
    (void)n_in; (void)d_ws; (void)ws_size;
    const float* pos = (const float*)d_in[0];
    const float* W1  = (const float*)d_in[1];
    const float* b1  = (const float*)d_in[2];
    const float* W2  = (const float*)d_in[3];
    const float* b2  = (const float*)d_in[4];
    const float* W3  = (const float*)d_in[5];
    const float* b3  = (const float*)d_in[6];
    float* out = (float*)d_out;

    const int Bn = in_sizes[0] / (N_PTS * DIMS);   // batch count (4)

    pairforce_zero_kernel<<<(out_size + BLOCK - 1) / BLOCK, BLOCK, 0, stream>>>(out, out_size);

    const int totalTiles = Bn * N_PTS * (N_PTS / TILE_J);  // 65536
    int blocks = 2048;                                     // 8192 waves -> 8 tiles/wave
    if (blocks * WAVES_PER_BLOCK > totalTiles)
        blocks = (totalTiles + WAVES_PER_BLOCK - 1) / WAVES_PER_BLOCK;

    pairforce_wmma_kernel<<<blocks, BLOCK, 0, stream>>>(pos, W1, b1, W2, b2, W3, b3, out, Bn);
}
